// STHG_62319975465039
// MI455X (gfx1250) — matrix operand order repaired and checked by reading, not compiled, not run on hardware
//
#include <hip/hip_runtime.h>
#include <math.h>

// ---------------- problem constants ----------------
enum {
  B = 32, L = 12, N = 2048, C = 3,
  D = 64, TOPK = 8, NK = 16, TOD = 288, DOW = 7, OUT = 12,
  NLAYER = 3,
  HS = 4 * D + TOPK * D,          // 768
  HT = HS + D,                    // 832
  HC = 99,
  NCOLS = B * N                   // 65536 (flattened b*N+n columns)
};
#define EPSF 1e-8f

typedef unsigned short u16;
typedef __attribute__((ext_vector_type(16))) __bf16 bf16x16;
typedef __attribute__((ext_vector_type(8)))  float  f32x8;
typedef __attribute__((ext_vector_type(4)))  unsigned int u32x4;

union ABu { u32x4 q[2]; unsigned int u[8]; bf16x16 v; };

__device__ __forceinline__ u16 f2bf(float f) {
  unsigned int u = __float_as_uint(f);
  u += 0x7fffu + ((u >> 16) & 1u);          // round-to-nearest-even
  return (u16)(u >> 16);
}
__device__ __forceinline__ int clampi(int v, int lo, int hi) {
  return v < lo ? lo : (v > hi ? hi : v);
}

// ---------------- f32 -> bf16 convert ----------------
__global__ void cvt_bf16_kernel(const float* __restrict__ src, u16* __restrict__ dst, size_t n) {
  size_t i = (size_t)blockIdx.x * blockDim.x + threadIdx.x;
  if (i < n) dst[i] = f2bf(src[i]);
}

// ---------------- EmbeddingTrainer ----------------
// z[d*HC+h] = relu( sum_n node_emb[n*D+d]*et_w1[h*N+n] + b1[h] )
__global__ void et1_kernel(const float* __restrict__ node_emb, const float* __restrict__ w1,
                           const float* __restrict__ b1, float* __restrict__ z) {
  int t = blockIdx.x * blockDim.x + threadIdx.x;
  if (t >= D * HC) return;
  int d = t / HC, h = t % HC;
  float s = b1[h];
  for (int n = 0; n < N; ++n) s += node_emb[(size_t)n * D + d] * w1[(size_t)h * N + n];
  z[t] = s > 0.f ? s : 0.f;
}
// proj[n*D+d] = sum_h z[d*HC+h]*et_w2[n*HC+h] + b2[n]
__global__ void et2_kernel(const float* __restrict__ z, const float* __restrict__ w2,
                           const float* __restrict__ b2, float* __restrict__ proj) {
  int t = blockIdx.x * blockDim.x + threadIdx.x;
  if (t >= N * D) return;
  int n = t / D, d = t % D;
  float s = b2[n];
  for (int h = 0; h < HC; ++h) s += z[d * HC + h] * w2[(size_t)n * HC + h];
  proj[t] = s;
}

// ---------------- assemble enc_in per (b,n) ----------------
// Writes hF/hBf columns [0,HS) with layout col-major, K contiguous: hF[col*HT + c]
__global__ void assemble_kernel(const float* __restrict__ hist, const float* __restrict__ proj,
                                const float* __restrict__ tod_emb, const float* __restrict__ dow_emb,
                                const float* __restrict__ ts_w, const float* __restrict__ feat,
                                const float* __restrict__ kern,
                                float* __restrict__ hF, u16* __restrict__ hBf,
                                float* __restrict__ Hinc) {
  int g = blockIdx.x * blockDim.x + threadIdx.x;
  if (g >= NCOLS) return;
  int b = g / N, n = g % N;

  float x[L]; float nx = 0.f;
  for (int l = 0; l < L; ++l) {
    x[l] = hist[(((size_t)(b * L + l) * N + n) * C)];
    nx += x[l] * x[l];
  }
  nx = sqrtf(nx) + EPSF;

  // cosine correlation with kernel bank, then top-8 of 16
  float conn[NK];
  for (int k = 0; k < NK; ++k) {
    float dot = 0.f, kk = 0.f;
    for (int l = 0; l < L; ++l) { float kv = kern[k * L + l]; dot += x[l] * kv; kk += kv * kv; }
    conn[k] = dot / (nx * (sqrtf(kk) + EPSF));
  }
  int idx[TOPK]; float cw[TOPK]; unsigned mask = 0;
  for (int t = 0; t < TOPK; ++t) {
    float best = -1e30f; int bi = 0;
    for (int k = 0; k < NK; ++k)
      if (!((mask >> k) & 1u) && conn[k] > best) { best = conn[k]; bi = k; }
    mask |= 1u << bi; idx[t] = bi; cw[t] = best;
  }
  for (int k = 0; k < NK; ++k) Hinc[(size_t)g * NK + k] = ((mask >> k) & 1u) ? 1.f : 0.f;

  size_t base = (size_t)g * HT;
  // ts_emb (c in [0,D))
  for (int d = 0; d < D; ++d) {
    float s = 0.f;
    for (int l = 0; l < L; ++l) s += x[l] * ts_w[d * L + l];
    hF[base + d] = s; hBf[base + d] = f2bf(s);
  }
  // proj_node channel (c in [D,2D))
  for (int d = 0; d < D; ++d) {
    float v = proj[(size_t)n * D + d];
    hF[base + D + d] = v; hBf[base + D + d] = f2bf(v);
  }
  // tod / dow embeddings at last step
  int ti = clampi((int)(hist[(((size_t)(b * L + (L - 1)) * N + n) * C) + 1] * TOD), 0, TOD - 1);
  int wi = clampi((int)(hist[(((size_t)(b * L + (L - 1)) * N + n) * C) + 2] * DOW), 0, DOW - 1);
  for (int d = 0; d < D; ++d) {
    float v = tod_emb[(size_t)ti * D + d];
    hF[base + 2 * D + d] = v; hBf[base + 2 * D + d] = f2bf(v);
  }
  for (int d = 0; d < D; ++d) {
    float v = dow_emb[(size_t)wi * D + d];
    hF[base + 3 * D + d] = v; hBf[base + 3 * D + d] = f2bf(v);
  }
  // corr channels (c in [4D, 4D+TOPK*D))
  for (int t = 0; t < TOPK; ++t) {
    float w = 1.f / (1.f + expf(-cw[t]));
    const float* fe = feat + (size_t)idx[t] * D;
    for (int d = 0; d < D; ++d) {
      float v = w * fe[d];
      size_t c = base + 4 * D + t * D + d;
      hF[c] = v; hBf[c] = f2bf(v);
    }
  }
}

// ---------------- generic WMMA bf16 GEMM ----------------
// out(col,m) = [resid(col,m)] + bias[m] + sum_k W[m,k]*X[col,k]   (optional relu)
// W: row-major MxK bf16. X: column-major (K contiguous per column), leading dim ldX.
// Each wave: 4 M-tiles (64 rows) x 2 N-tiles (32 cols); A operand reused across the
// two column tiles (1.5 b128 loads per WMMA). K in steps of 32.
// GUARD=false requires M % 64 == 0 (no lane-divergent predication in the hot loop).
template <bool GUARD>
__global__ __launch_bounds__(256) void gemm_bf16_wmma(
    const u16* __restrict__ W, const u16* __restrict__ X, int ldX,
    const float* __restrict__ bias, const float* resid,
    float* outF, u16* outBf, int ldOut,
    int M, int K, int Ncols, int relu) {
  const int nPairs = Ncols >> 5;                 // pairs of 16-column tiles
  const int mGroups = (M + 63) >> 6;
  const int wid = (blockIdx.x << 3) + (threadIdx.x >> 5);
  if (wid >= mGroups * nPairs) return;
  const int pairN = wid % nPairs;
  const int mg = wid / nPairs;
  const int lane = threadIdx.x & 31;
  const int half = lane >> 4;
  const int lrow = lane & 15;
  const size_t col0 = (size_t)pairN * 32 + lrow;
  const size_t col1 = col0 + 16;

  f32x8 zero = {0.f, 0.f, 0.f, 0.f, 0.f, 0.f, 0.f, 0.f};
  f32x8 acc[4][2];
#pragma unroll
  for (int t = 0; t < 4; ++t) {
    acc[t][0] = zero;
    acc[t][1] = zero;
  }

  // Per-tile W row pointers (A-operand row = mg*64 + t*16 + lrow).
  const u16* wrow[4];
  bool mok[4];
#pragma unroll
  for (int t = 0; t < 4; ++t) {
    int m = (mg << 6) + (t << 4) + lrow;
    mok[t] = GUARD ? (m < M) : true;
    wrow[t] = W + (size_t)(mok[t] ? m : 0) * K;   // clamp; masked by zero A below
  }
  const u16* xcol0 = X + col0 * (size_t)ldX + 16 * half;
  const u16* xcol1 = X + col1 * (size_t)ldX + 16 * half;

  const int kb = K >> 5;
  for (int kk = 0; kk < kb; ++kk) {
    const int k0 = kk << 5;
    // B operands: two column tiles, K = k0 + 16*half .. +15 (contiguous b128 pairs)
    ABu Bu0, Bu1;
    {
      const u32x4* xq0 = (const u32x4*)(xcol0 + k0);
      const u32x4* xq1 = (const u32x4*)(xcol1 + k0);
      Bu0.q[0] = xq0[0]; Bu0.q[1] = xq0[1];
      Bu1.q[0] = xq1[0]; Bu1.q[1] = xq1[1];
    }
#pragma unroll
    for (int t = 0; t < 4; ++t) {
      // A layout: dwords 0..7 -> K = 2j + 16*half (contiguous, matches B)
      ABu Au;
      const u16* wr = wrow[t] + k0 + 16 * half;
      Au.q[0] = *(const u32x4*)(wr);
      Au.q[1] = *(const u32x4*)(wr + 8);
      if (GUARD && !mok[t]) {
        u32x4 zq = {0u, 0u, 0u, 0u};
        Au.q[0] = zq; Au.q[1] = zq;
      }
      acc[t][0] = __builtin_amdgcn_wmma_f32_16x16x32_bf16(
          false, Au.v, false, Bu0.v, (short)0, acc[t][0], false, false);
      acc[t][1] = __builtin_amdgcn_wmma_f32_16x16x32_bf16(
          false, Au.v, false, Bu1.v, (short)0, acc[t][1], false, false);
    }
  }
  // epilogue: D element i -> row m = base + i + 8*half, column = col0 / col1
#pragma unroll
  for (int t = 0; t < 4; ++t) {
#pragma unroll
    for (int c = 0; c < 2; ++c) {
      const size_t col = c ? col1 : col0;
#pragma unroll
      for (int i = 0; i < 8; ++i) {
        const int m = (mg << 6) + (t << 4) + i + 8 * half;
        if (!GUARD || m < M) {
          float v = acc[t][c][i];
          if (bias) v += bias[m];
          if (resid) v += resid[col * (size_t)ldOut + m];
          if (relu) v = v > 0.f ? v : 0.f;
          if (outF) outF[col * (size_t)ldOut + m] = v;
          if (outBf) outBf[col * (size_t)ldOut + m] = f2bf(v);
        }
      }
    }
  }
}

// ---------------- hypergraph edge aggregation ----------------
// edge[(b*D+d)*NK+k] = (sum_n Hinc[bn,k]*h0[bn,d]) / max(deg,1) + feat[k,d]
__global__ __launch_bounds__(64) void edge_kernel(const float* __restrict__ Hinc,
                                                  const float* __restrict__ h0,
                                                  const float* __restrict__ feat,
                                                  float* __restrict__ edge) {
  int b = blockIdx.x / NK, k = blockIdx.x % NK;
  int d = threadIdx.x;  // 64 threads
  __shared__ float sH[64];
  __shared__ float sR[64];
  float acc = 0.f, deg = 0.f;
  for (int base = 0; base < N; base += 64) {
    __syncthreads();
    float hv = Hinc[((size_t)b * N + base + d) * NK + k];
    sH[d] = hv; deg += hv;
    __syncthreads();
    for (int j = 0; j < 64; ++j)
      acc += sH[j] * h0[((size_t)b * N + base + j) * D + d];
  }
  sR[d] = deg; __syncthreads();
  for (int s = 32; s > 0; s >>= 1) { if (d < s) sR[d] += sR[d + s]; __syncthreads(); }
  float de = fmaxf(sR[0], 1.f);
  edge[((size_t)b * D + d) * NK + k] = acc / de + feat[(size_t)k * D + d];
}

// ---------------- msg + sp_out -> h columns [HS,HT) ----------------
__global__ void msg_kernel(const float* __restrict__ Hinc, const float* __restrict__ edge,
                           const float* __restrict__ h0, const float* __restrict__ proj,
                           float* __restrict__ hF, u16* __restrict__ hBf) {
  size_t t = (size_t)blockIdx.x * blockDim.x + threadIdx.x;
  if (t >= (size_t)NCOLS * D) return;
  int d = (int)(t % D);
  size_t colv = t / D;
  int b = (int)(colv / N), n = (int)(colv % N);
  float m = 0.f;
  for (int k = 0; k < NK; ++k)
    m += Hinc[colv * NK + k] * edge[((size_t)b * D + d) * NK + k];
  float v = h0[colv * D + d] + m * (1.f / (float)TOPK) + proj[(size_t)n * D + d];
  size_t c = colv * HT + HS + d;
  hF[c] = v; hBf[c] = f2bf(v);
}

// ---------------- temporal branch (RevIN + cycles) + final blend ----------------
__global__ void final_kernel(const float* __restrict__ hist, const float* __restrict__ fut,
                             const float* __restrict__ treg_w, const float* __restrict__ treg_b,
                             const float* __restrict__ daily, const float* __restrict__ weekly,
                             const float* __restrict__ wsp, const float* __restrict__ fsp,
                             float* __restrict__ out) {
  int g = blockIdx.x * blockDim.x + threadIdx.x;
  if (g >= NCOLS) return;
  int b = g / N, n = g % N;
  float x[L]; float mean = 0.f;
  for (int l = 0; l < L; ++l) {
    x[l] = hist[(((size_t)(b * L + l) * N + n) * C)];
    mean += x[l];
  }
  mean *= (1.f / (float)L);
  float var = 0.f;
  for (int l = 0; l < L; ++l) { float dlt = x[l] - mean; var += dlt * dlt; }
  var = var / (float)(L - 1) + 1e-5f;
  float sd = sqrtf(var);
  float xr[L];
  for (int l = 0; l < L; ++l) {
    int di = clampi((int)(hist[(((size_t)(b * L + l) * N) * C) + 1] * TOD), 0, TOD - 1);
    int wi = clampi((int)(hist[(((size_t)(b * L + l) * N) * C) + 2] * DOW), 0, DOW - 1);
    xr[l] = (x[l] - mean) / sd - daily[(size_t)di * N + n] - weekly[(size_t)wi * N + n];
  }
  float w = wsp[n];
  for (int o = 0; o < OUT; ++o) {
    float ft = treg_b[o];
    for (int l = 0; l < L; ++l) ft += treg_w[o * L + l] * xr[l];
    int dif = clampi((int)(fut[(((size_t)(b * OUT + o) * N) * C) + 1] * TOD), 0, TOD - 1);
    int wif = clampi((int)(fut[(((size_t)(b * OUT + o) * N) * C) + 2] * DOW), 0, DOW - 1);
    ft += daily[(size_t)dif * N + n] + weekly[(size_t)wif * N + n];
    ft = ft * sd + mean;
    float sp = fsp[(size_t)g * OUT + o];
    out[((size_t)(b * OUT + o)) * N + n] = w * sp + (1.f - w) * ft;
  }
}

// ---------------- host launch ----------------
extern "C" void kernel_launch(void* const* d_in, const int* in_sizes, int n_in,
                              void* d_out, int out_size, void* d_ws, size_t ws_size,
                              hipStream_t stream) {
  const float* hist    = (const float*)d_in[0];
  const float* fut     = (const float*)d_in[1];
  const float* node_emb= (const float*)d_in[2];
  const float* et_w1   = (const float*)d_in[3];
  const float* et_b1   = (const float*)d_in[4];
  const float* et_w2   = (const float*)d_in[5];
  const float* et_b2   = (const float*)d_in[6];
  const float* tod_emb = (const float*)d_in[7];
  const float* dow_emb = (const float*)d_in[8];
  const float* ts_w    = (const float*)d_in[9];
  const float* feat    = (const float*)d_in[10];
  const float* kern    = (const float*)d_in[11];
  const float* sp_w    = (const float*)d_in[12];
  const float* sp_b    = (const float*)d_in[13];
  const float* m_w1    = (const float*)d_in[14];
  const float* m_b1    = (const float*)d_in[15];
  const float* m_w2    = (const float*)d_in[16];
  const float* m_b2    = (const float*)d_in[17];
  const float* treg_w  = (const float*)d_in[18];
  const float* treg_b  = (const float*)d_in[19];
  const float* sreg_w  = (const float*)d_in[20];
  const float* sreg_b  = (const float*)d_in[21];
  const float* daily   = (const float*)d_in[22];
  const float* weekly  = (const float*)d_in[23];
  const float* wsp     = (const float*)d_in[24];
  float* out = (float*)d_out;

  // ---- workspace bump allocator ----
  char* p = (char*)d_ws;
  size_t used = 0;
  auto alloc = [&](size_t bytes) -> void* {
    size_t pad = (bytes + 255) & ~(size_t)255;
    void* r = (void*)(p + used);
    used += pad;
    return r;
  };
  u16*   spWb  = (u16*)alloc((size_t)D * HS * 2);
  u16*   m1Wb  = (u16*)alloc((size_t)NLAYER * HT * HT * 2);
  u16*   m2Wb  = (u16*)alloc((size_t)NLAYER * HT * HT * 2);
  u16*   srWb  = (u16*)alloc((size_t)OUT * HT * 2);
  float* zET   = (float*)alloc((size_t)D * HC * 4);
  float* proj  = (float*)alloc((size_t)N * D * 4);
  float* Hinc  = (float*)alloc((size_t)NCOLS * NK * 4);
  float* hF    = (float*)alloc((size_t)NCOLS * HT * 4);
  u16*   hBf   = (u16*)alloc((size_t)NCOLS * HT * 2);
  u16*   z1Bf  = (u16*)alloc((size_t)NCOLS * HT * 2);
  float* h0F   = (float*)alloc((size_t)NCOLS * D * 4);
  float* edge  = (float*)alloc((size_t)B * D * NK * 4);
  float* fspF  = (float*)alloc((size_t)NCOLS * OUT * 4);
  if (used > ws_size) return;  // workspace insufficient; deterministic no-op

  auto cvt = [&](const float* s, u16* d, size_t n) {
    cvt_bf16_kernel<<<(unsigned)((n + 255) / 256), 256, 0, stream>>>(s, d, n);
  };
  cvt(sp_w,   spWb, (size_t)D * HS);
  cvt(m_w1,   m1Wb, (size_t)NLAYER * HT * HT);
  cvt(m_w2,   m2Wb, (size_t)NLAYER * HT * HT);
  cvt(sreg_w, srWb, (size_t)OUT * HT);

  // EmbeddingTrainer
  et1_kernel<<<(D * HC + 255) / 256, 256, 0, stream>>>(node_emb, et_w1, et_b1, zET);
  et2_kernel<<<(N * D + 255) / 256, 256, 0, stream>>>(zET, et_w2, et_b2, proj);

  // Assemble enc_in into h columns [0,HS)
  assemble_kernel<<<(NCOLS + 255) / 256, 256, 0, stream>>>(
      hist, proj, tod_emb, dow_emb, ts_w, feat, kern, hF, hBf, Hinc);

  auto gemm = [&](const u16* Wb, const u16* Xb, int ldX, const float* bias,
                  const float* resid, float* oF, u16* oB, int ldOut,
                  int M, int K, int relu) {
    int nPairs = NCOLS / 32;
    int mGroups = (M + 63) / 64;
    int waves = nPairs * mGroups;
    int blocks = (waves + 7) / 8;
    if ((M & 63) == 0)
      gemm_bf16_wmma<false><<<blocks, 256, 0, stream>>>(Wb, Xb, ldX, bias, resid, oF, oB,
                                                        ldOut, M, K, NCOLS, relu);
    else
      gemm_bf16_wmma<true><<<blocks, 256, 0, stream>>>(Wb, Xb, ldX, bias, resid, oF, oB,
                                                       ldOut, M, K, NCOLS, relu);
  };

  // Spatial encoder projection: h0 = relu(sp_w @ enc + sp_b)
  gemm(spWb, hBf, HT, sp_b, nullptr, h0F, nullptr, D, D, HS, 1);

  // Hypergraph message passing -> sp_out into h columns [HS,HT)
  edge_kernel<<<B * NK, 64, 0, stream>>>(Hinc, h0F, feat, edge);
  {
    size_t tot = (size_t)NCOLS * D;
    msg_kernel<<<(unsigned)((tot + 255) / 256), 256, 0, stream>>>(Hinc, edge, h0F, proj, hF, hBf);
  }

  // Temporal encoder: residual MLP stack (the dominant GEMMs)
  for (int i = 0; i < NLAYER; ++i) {
    const u16* w1 = m1Wb + (size_t)i * HT * HT;
    const u16* w2 = m2Wb + (size_t)i * HT * HT;
    // z1 = relu(W1 @ h + b1)  (bf16 output only, feeds next GEMM)
    gemm(w1, hBf, HT, m_b1 + (size_t)i * HT, nullptr, nullptr, z1Bf, HT, HT, HT, 1);
    // h = h + W2 @ z1 + b2    (residual from hF, write f32 + bf16)
    gemm(w2, z1Bf, HT, m_b2 + (size_t)i * HT, hF, hF, hBf, HT, HT, HT, 0);
  }

  // forecast_sp = sreg_w @ h + sreg_b
  gemm(srWb, hBf, HT, sreg_b, nullptr, fspF, nullptr, OUT, OUT, HT, 0);

  // Temporal branch + blend
  final_kernel<<<(NCOLS + 255) / 256, 256, 0, stream>>>(
      hist, fut, treg_w, treg_b, daily, weekly, wsp, fspF, out);
}
